// PhaseFieldPredictor_7404523619175
// MI455X (gfx1250) — matrix operand order repaired and checked by reading, not compiled, not run on hardware
//
#include <hip/hip_runtime.h>
#include <hip/hip_bf16.h>

typedef __attribute__((ext_vector_type(16))) _Float16 v16h;
typedef __attribute__((ext_vector_type(8)))  float    v8f;

#define GRID_G 256
#define NN     65536
#define TT     5
#define CC     10
#define HHID   32
#define NGATE  128
#define WIDTH  64
#define BB     2
#define KW     32
#define OUTC   10
#define NROWS  (BB*NN)   // 131072 feature rows / sequences

// Branch-free transcendentals (gate pre-activations are O(1) here).
__device__ __forceinline__ float fast_rcp(float x) {
#if __has_builtin(__builtin_amdgcn_rcpf)
  return __builtin_amdgcn_rcpf(x);
#else
  return 1.0f / x;
#endif
}
__device__ __forceinline__ float sigm(float x) { return fast_rcp(1.0f + __expf(-x)); }
__device__ __forceinline__ float fast_tanh(float x) {
#if __has_builtin(__builtin_amdgcn_tanhf)
  return __builtin_amdgcn_tanhf(x);
#else
  return 1.0f - 2.0f * fast_rcp(__expf(2.0f * x) + 1.0f);
#endif
}

__device__ __forceinline__ v8f wmma_f16(v16h a, v16h b, v8f c) {
  return __builtin_amdgcn_wmma_f32_16x16x32_f16(false, a, false, b, (short)0, c, false, false);
}

// K-index base for element pair i (0..7) of a 16x32 f16 A-fragment (contiguous k order of this file, same as the B-frags)
__device__ __forceinline__ int afrag_kbase(int i, int half) {
  return half*16 + 2*i;
}

// ---- unified LDS arena for kernel 1 (aliasing keeps B-frags from being pinned) ----
#define OFF_WB   0                                  // 4 mats x 8 tiles x 32 lanes x 16 halves (f16)
#define OFF_FB   (OFF_WB + 4*8*32*16*2)             // 32768
#define OFF_BIAS (OFF_FB + 4*32*16*2)               // +4096
#define OFF_XS   (OFF_BIAS + 2*NGATE*4)             // +1024   xs[32][128] f32 (ch 10..31 zero)
#define OFF_HX   (OFF_XS + 32*128*4)                // +16384  hx[8][16][33] f32
#define SMEM_SZ  (OFF_HX + 8*16*33*4)               // ~71 KB

// ---------------------------------------------------------------------------
// Kernel 1: per-node 2-layer LSTM over T steps + fc1, all GEMMs via WMMA f16.
// Each wave owns 16 sequences (rows); gates = 8 N-tiles of 16.
// ---------------------------------------------------------------------------
__global__ __launch_bounds__(256)
__attribute__((amdgpu_waves_per_eu(4)))
void lstm_fc1_kernel(
    const float* __restrict__ x,
    const float* __restrict__ Wih0, const float* __restrict__ Whh0,
    const float* __restrict__ bih0, const float* __restrict__ bhh0,
    const float* __restrict__ Wih1, const float* __restrict__ Whh1,
    const float* __restrict__ bih1, const float* __restrict__ bhh1,
    const float* __restrict__ fc1_w, const float* __restrict__ fc1_b,
    float* __restrict__ feats)
{
  __shared__ __align__(32) unsigned char smem[SMEM_SZ];
  _Float16* wBp   = (_Float16*)(smem + OFF_WB);
  _Float16* fBp   = (_Float16*)(smem + OFF_FB);
  float*    biasS = (float*)(smem + OFF_BIAS);
  float*    xs    = (float*)(smem + OFF_XS);
  float*    hx    = (float*)(smem + OFF_HX);

  // ---- stage weights as B-fragments, biases, zero x padding ----
  for (int idx = threadIdx.x; idx < 4*8*32*16; idx += 256) {
    int mat  = idx >> 12;
    int tile = (idx >> 9) & 7;
    int lane = (idx >> 4) & 31;
    int j    = idx & 15;
    int n  = tile*16 + (lane & 15);
    int kk = ((lane >> 4) << 4) + j;
    float v;
    if      (mat == 0) v = (kk < CC) ? Wih0[n*CC + kk] : 0.0f;   // pad K 10->32
    else if (mat == 1) v = Whh0[n*HHID + kk];
    else if (mat == 2) v = Wih1[n*HHID + kk];
    else               v = Whh1[n*HHID + kk];
    wBp[idx] = (_Float16)v;
  }
  for (int idx = threadIdx.x; idx < 4*32*16; idx += 256) {
    int tile = idx >> 9;
    int lane = (idx >> 4) & 31;
    int j    = idx & 15;
    int n  = tile*16 + (lane & 15);
    int kk = ((lane >> 4) << 4) + j;
    fBp[idx] = (_Float16)fc1_w[n*HHID + kk];
  }
  for (int n = threadIdx.x; n < NGATE; n += 256) {
    biasS[n]         = bih0[n] + bhh0[n];
    biasS[NGATE + n] = bih1[n] + bhh1[n];
  }
  for (int idx = threadIdx.x; idx < 32*128; idx += 256) xs[idx] = 0.0f;
  __syncthreads();

  int wave = threadIdx.x >> 5;
  int lane = threadIdx.x & 31;
  int col  = lane & 15;
  int half = lane >> 4;
  long blockBase = (long)blockIdx.x * 128;
  long base  = blockBase + wave * 16;
  int  lrow  = wave*16 + col;           // local A-fragment row for this lane

  v16h h0a, h1a;
  v8f  c0a, c0b, c1a, c1b;
#pragma unroll
  for (int i = 0; i < 16; ++i) { h0a[i] = (_Float16)0.0f; h1a[i] = (_Float16)0.0f; }
#pragma unroll
  for (int r = 0; r < 8; ++r) { c0a[r] = 0.f; c0b[r] = 0.f; c1a[r] = 0.f; c1b[r] = 0.f; }

  for (int t = 0; t < TT; ++t) {
    // ---- coalesced stage of x[:, t, :, block rows] into LDS (ch 10..31 stay 0) ----
    __syncthreads();   // previous iteration's xs reads are done
    for (int idx = threadIdx.x; idx < 128*CC; idx += 256) {
      int ch = idx >> 7;
      int rr = idx & 127;
      long row = blockBase + rr;
      int b    = (int)(row >> 16);
      int node = (int)(row & (NN - 1));
      xs[ch*128 + rr] = x[((long)(b*TT + t)*CC + ch)*NN + node];
    }
    __syncthreads();

    // ---- x_t A-fragment from LDS (K padded to 32) ----
    v16h xa;
#pragma unroll
    for (int i = 0; i < 8; ++i) {
      int kb = afrag_kbase(i, half);
      xa[2*i]   = (_Float16)xs[kb*128 + lrow];
      xa[2*i+1] = (_Float16)xs[(kb+1)*128 + lrow];
    }

    // ---- layer 0: z = x@Wih0^T + h0@Whh0^T + bias ----
    v8f d[8];
#pragma unroll
    for (int tl = 0; tl < 8; ++tl) {
      float bv = biasS[tl*16 + col];
#pragma unroll
      for (int r = 0; r < 8; ++r) d[tl][r] = bv;
      d[tl] = wmma_f16(xa,  *(const v16h*)&wBp[((0*8 + tl)*32 + lane)*16], d[tl]);
      d[tl] = wmma_f16(h0a, *(const v16h*)&wBp[((1*8 + tl)*32 + lane)*16], d[tl]);
    }
#pragma unroll
    for (int j2 = 0; j2 < 2; ++j2) {
#pragma unroll
      for (int r = 0; r < 8; ++r) {
        float iv = d[0+j2][r], fv = d[2+j2][r], gv = d[4+j2][r], ov = d[6+j2][r];
        float cc = j2 ? c0b[r] : c0a[r];
        cc = sigm(fv)*cc + sigm(iv)*fast_tanh(gv);
        if (j2) c0b[r] = cc; else c0a[r] = cc;
        hx[(wave*16 + half*8 + r)*33 + j2*16 + col] = sigm(ov)*fast_tanh(cc);
      }
    }
    // wave-private LDS staging: DS ops are in-order per wave, no barrier needed
#pragma unroll
    for (int i = 0; i < 8; ++i) {
      int kb = afrag_kbase(i, half);
      h0a[2*i]   = (_Float16)hx[(wave*16 + col)*33 + kb];
      h0a[2*i+1] = (_Float16)hx[(wave*16 + col)*33 + kb + 1];
    }

    // ---- layer 1: z = h0@Wih1^T + h1@Whh1^T + bias ----
#pragma unroll
    for (int tl = 0; tl < 8; ++tl) {
      float bv = biasS[NGATE + tl*16 + col];
#pragma unroll
      for (int r = 0; r < 8; ++r) d[tl][r] = bv;
      d[tl] = wmma_f16(h0a, *(const v16h*)&wBp[((2*8 + tl)*32 + lane)*16], d[tl]);
      d[tl] = wmma_f16(h1a, *(const v16h*)&wBp[((3*8 + tl)*32 + lane)*16], d[tl]);
    }
#pragma unroll
    for (int j2 = 0; j2 < 2; ++j2) {
#pragma unroll
      for (int r = 0; r < 8; ++r) {
        float iv = d[0+j2][r], fv = d[2+j2][r], gv = d[4+j2][r], ov = d[6+j2][r];
        float cc = j2 ? c1b[r] : c1a[r];
        cc = sigm(fv)*cc + sigm(iv)*fast_tanh(gv);
        if (j2) c1b[r] = cc; else c1a[r] = cc;
        hx[(wave*16 + half*8 + r)*33 + j2*16 + col] = sigm(ov)*fast_tanh(cc);
      }
    }
#pragma unroll
    for (int i = 0; i < 8; ++i) {
      int kb = afrag_kbase(i, half);
      h1a[2*i]   = (_Float16)hx[(wave*16 + col)*33 + kb];
      h1a[2*i+1] = (_Float16)hx[(wave*16 + col)*33 + kb + 1];
    }
  }

  // ---- fc1: feats = relu(h1 @ fc1_w^T + b), N = 64 (4 tiles), K = 32 ----
#pragma unroll
  for (int tl = 0; tl < 4; ++tl) {
    float bv = fc1_b[tl*16 + col];
    v8f f;
#pragma unroll
    for (int r = 0; r < 8; ++r) f[r] = bv;
    f = wmma_f16(h1a, *(const v16h*)&fBp[(tl*32 + lane)*16], f);
#pragma unroll
    for (int r = 0; r < 8; ++r) {
      long row = base + half*8 + r;
      feats[row*WIDTH + tl*16 + col] = fmaxf(f[r], 0.0f);
    }
  }
}

// ---------------------------------------------------------------------------
// Kernel 2: Y = feats @ conv_w[k]   (131072x64 @ 64x64), WMMA f16, K=64.
// ---------------------------------------------------------------------------
__global__ __launch_bounds__(256) void conv_gemm_kernel(
    const float* __restrict__ feats, const float* __restrict__ Wc,
    float* __restrict__ Y)
{
  __shared__ __align__(32) _Float16 cB[4][2][32][16];  // [n-tile][k-block][lane][16]
  for (int idx = threadIdx.x; idx < 4*2*32*16; idx += 256) {
    int tile = idx >> 10;
    int kb   = (idx >> 9) & 1;
    int lane = (idx >> 4) & 31;
    int j    = idx & 15;
    int n  = tile*16 + (lane & 15);
    int kk = kb*32 + ((lane >> 4) << 4) + j;
    cB[tile][kb][lane][j] = (_Float16)Wc[kk*WIDTH + n];
  }
  __syncthreads();

  int wave = threadIdx.x >> 5, lane = threadIdx.x & 31;
  int col = lane & 15, half = lane >> 4;
  long base = ((long)blockIdx.x * 8 + wave) * 16;
  const float* fp = feats + (base + col) * WIDTH;

  v16h a0, a1;
#pragma unroll
  for (int i = 0; i < 8; ++i) {
    int kb = afrag_kbase(i, half);
    float2 u = *(const float2*)(fp + kb);
    float2 v = *(const float2*)(fp + 32 + kb);
    a0[2*i] = (_Float16)u.x; a0[2*i+1] = (_Float16)u.y;
    a1[2*i] = (_Float16)v.x; a1[2*i+1] = (_Float16)v.y;
  }
#pragma unroll
  for (int tl = 0; tl < 4; ++tl) {
    v8f d;
#pragma unroll
    for (int r = 0; r < 8; ++r) d[r] = 0.0f;
    d = wmma_f16(a0, *(const v16h*)&cB[tl][0][lane][0], d);
    d = wmma_f16(a1, *(const v16h*)&cB[tl][1][lane][0], d);
#pragma unroll
    for (int r = 0; r < 8; ++r)
      Y[(base + half*8 + r)*WIDTH + tl*16 + col] = d[r];
  }
}

// ---------------------------------------------------------------------------
// Kernel 3: graph conv as 3x3 stencil: feats = [relu](Y + sum_nbr w*Y_nbr + b)
// (8-neighbor gaussian-gated scatter-add == 2-weight stencil; dist^2 in {1,2})
// ---------------------------------------------------------------------------
__global__ __launch_bounds__(256) void stencil_kernel(
    const float* __restrict__ Y, const float* __restrict__ conv_b_k,
    const float* __restrict__ gparam, int k, int do_relu,
    float* __restrict__ feats)
{
  int tid  = blockIdx.x * 256 + threadIdx.x;  // BB*NN*16 threads (float4 of channels)
  int cg   = tid & 15;
  int node = (tid >> 4) & (NN - 1);
  int b    = tid >> 20;
  int i = node >> 8, j = node & (GRID_G - 1);

  float gp  = gparam[k];
  float inv = 1.0f / (gp*gp + 1e-8f);
  float w1 = __expf(-inv);         // straight neighbors (dist^2 = 1)
  float w2 = __expf(-2.0f * inv);  // diagonal neighbors (dist^2 = 2)

  const float4* Yb = (const float4*)Y + (long)b * NN * 16;
  float4 acc  = Yb[(long)node * 16 + cg];
  float4 bias = ((const float4*)conv_b_k)[cg];
  acc.x += bias.x; acc.y += bias.y; acc.z += bias.z; acc.w += bias.w;

#pragma unroll
  for (int di = -1; di <= 1; ++di) {
#pragma unroll
    for (int dj = -1; dj <= 1; ++dj) {
      if (di == 0 && dj == 0) continue;
      int ni = i + di, nj = j + dj;
      if (ni < 0 || ni >= GRID_G || nj < 0 || nj >= GRID_G) continue;
      float w = (di != 0 && dj != 0) ? w2 : w1;
      float4 v = Yb[(long)(ni * GRID_G + nj) * 16 + cg];
      acc.x += w * v.x; acc.y += w * v.y; acc.z += w * v.z; acc.w += w * v.w;
    }
  }
  if (do_relu) {
    acc.x = fmaxf(acc.x, 0.f); acc.y = fmaxf(acc.y, 0.f);
    acc.z = fmaxf(acc.z, 0.f); acc.w = fmaxf(acc.w, 0.f);
  }
  ((float4*)feats)[((long)b * NN + node) * 16 + cg] = acc;
}

// ---------------------------------------------------------------------------
// Kernel 4: head: out = relu(feats@fc2^T+b2)@fc3^T+b3, transposed to (B,10,H,W)
// ---------------------------------------------------------------------------
__global__ __launch_bounds__(256) void head_kernel(
    const float* __restrict__ feats,
    const float* __restrict__ fc2_w, const float* __restrict__ fc2_b,
    const float* __restrict__ fc3_w, const float* __restrict__ fc3_b,
    float* __restrict__ out)
{
  __shared__ __align__(32) _Float16 w2B[2][2][32][16];  // N=32 (2 tiles), K=64 (2 blocks)
  __shared__ __align__(32) _Float16 w3B[32][16];        // N=16 (pad from 10), K=32
  __shared__ float hx[8][16][33];

  for (int idx = threadIdx.x; idx < 2*2*32*16; idx += 256) {
    int tile = idx >> 10;
    int kb   = (idx >> 9) & 1;
    int lane = (idx >> 4) & 31;
    int j    = idx & 15;
    int n  = tile*16 + (lane & 15);
    int kk = kb*32 + ((lane >> 4) << 4) + j;
    w2B[tile][kb][lane][j] = (_Float16)fc2_w[n*WIDTH + kk];
  }
  for (int idx = threadIdx.x; idx < 32*16; idx += 256) {
    int lane = idx >> 4;
    int j    = idx & 15;
    int n  = lane & 15;
    int kk = ((lane >> 4) << 4) + j;
    w3B[lane][j] = (n < OUTC) ? (_Float16)fc3_w[n*KW + kk] : (_Float16)0.0f;
  }
  __syncthreads();

  int wave = threadIdx.x >> 5, lane = threadIdx.x & 31;
  int col = lane & 15, half = lane >> 4;
  long base = ((long)blockIdx.x * 8 + wave) * 16;
  const float* fp = feats + (base + col) * WIDTH;

  v16h a0, a1;
#pragma unroll
  for (int i = 0; i < 8; ++i) {
    int kb = afrag_kbase(i, half);
    float2 u = *(const float2*)(fp + kb);
    float2 v = *(const float2*)(fp + 32 + kb);
    a0[2*i] = (_Float16)u.x; a0[2*i+1] = (_Float16)u.y;
    a1[2*i] = (_Float16)v.x; a1[2*i+1] = (_Float16)v.y;
  }
  // fc2 + relu, stage result through LDS (C-layout -> A-layout)
#pragma unroll
  for (int tl = 0; tl < 2; ++tl) {
    float bv = fc2_b[tl*16 + col];
    v8f d;
#pragma unroll
    for (int r = 0; r < 8; ++r) d[r] = bv;
    d = wmma_f16(a0, *(const v16h*)&w2B[tl][0][lane][0], d);
    d = wmma_f16(a1, *(const v16h*)&w2B[tl][1][lane][0], d);
#pragma unroll
    for (int r = 0; r < 8; ++r)
      hx[wave][half*8 + r][tl*16 + col] = fmaxf(d[r], 0.0f);
  }
  v16h ra;
#pragma unroll
  for (int i = 0; i < 8; ++i) {
    int kb = afrag_kbase(i, half);
    ra[2*i]   = (_Float16)hx[wave][col][kb];
    ra[2*i+1] = (_Float16)hx[wave][col][kb + 1];
  }
  // fc3 (N padded 10 -> 16)
  float b3 = (col < OUTC) ? fc3_b[col] : 0.0f;
  v8f o;
#pragma unroll
  for (int r = 0; r < 8; ++r) o[r] = b3;
  o = wmma_f16(ra, *(const v16h*)&w3B[lane][0], o);

  if (col < OUTC) {
#pragma unroll
    for (int r = 0; r < 8; ++r) {
      long row  = base + half*8 + r;
      int  b    = (int)(row >> 16);
      int  node = (int)(row & (NN - 1));
      out[((long)b * OUTC + col) * NN + node] = o[r];
    }
  }
}

// ---------------------------------------------------------------------------
extern "C" void kernel_launch(void* const* d_in, const int* in_sizes, int n_in,
                              void* d_out, int out_size, void* d_ws, size_t ws_size,
                              hipStream_t stream) {
  (void)in_sizes; (void)n_in; (void)out_size; (void)ws_size;
  const float* x      = (const float*)d_in[0];
  // d_in[1..3]: edge_src/tgt/attr — replaced by exact 3x3 stencil equivalence
  const float* Wih0   = (const float*)d_in[4];
  const float* Whh0   = (const float*)d_in[5];
  const float* bih0   = (const float*)d_in[6];
  const float* bhh0   = (const float*)d_in[7];
  const float* Wih1   = (const float*)d_in[8];
  const float* Whh1   = (const float*)d_in[9];
  const float* bih1   = (const float*)d_in[10];
  const float* bhh1   = (const float*)d_in[11];
  const float* fc1_w  = (const float*)d_in[12];
  const float* fc1_b  = (const float*)d_in[13];
  const float* conv_w = (const float*)d_in[14];
  const float* conv_b = (const float*)d_in[15];
  const float* gparam = (const float*)d_in[16];
  const float* fc2_w  = (const float*)d_in[17];
  const float* fc2_b  = (const float*)d_in[18];
  const float* fc3_w  = (const float*)d_in[19];
  const float* fc3_b  = (const float*)d_in[20];
  float* out = (float*)d_out;

  float* feats = (float*)d_ws;                       // 131072 x 64 f32 (32 MB)
  float* Y     = feats + (size_t)NROWS * WIDTH;      // 131072 x 64 f32 (32 MB)

  dim3 blk(256);
  lstm_fc1_kernel<<<NROWS/128, blk, 0, stream>>>(
      x, Wih0, Whh0, bih0, bhh0, Wih1, Whh1, bih1, bhh1, fc1_w, fc1_b, feats);

  for (int k = 0; k < 4; ++k) {
    conv_gemm_kernel<<<NROWS/128, blk, 0, stream>>>(feats, conv_w + k*WIDTH*WIDTH, Y);
    stencil_kernel<<<(BB*NN*16)/256, blk, 0, stream>>>(
        Y, conv_b + k*WIDTH, gparam, k, (k != 3) ? 1 : 0, feats);
  }

  head_kernel<<<NROWS/128, blk, 0, stream>>>(feats, fc2_w, fc2_b, fc3_w, fc3_b, out);
}
